// GINEModel_26585847562989
// MI455X (gfx1250) — matrix operand order repaired and checked by reading, not compiled, not run on hardware
//
#include <hip/hip_runtime.h>

// ---------------------------------------------------------------------------
// GINE model for MI455X (gfx1250). GEMMs run on the WMMA pipe
// (v_wmma_f32_16x16x32_bf16, f32 accumulate), edge phase uses L2 f32 atomics.
// Staging loads are branch-free (clamped addresses + cndmask) and all hot-path
// index math is unsigned 32-bit to avoid 64-bit multiply chains.
// ---------------------------------------------------------------------------

typedef __attribute__((ext_vector_type(16))) __bf16 v16bf;
typedef __attribute__((ext_vector_type(8)))  float  v8f;

union FragU { v16bf v; unsigned int u[8]; };

__device__ __forceinline__ unsigned int pack2bf(float a, float b) {
    // round-to-nearest-even f32 -> bf16, pack low|high
    unsigned int ua = __float_as_uint(a);
    unsigned int ub = __float_as_uint(b);
    ua = (ua + 0x7FFFu + ((ua >> 16) & 1u)) >> 16;
    ub = (ub + 0x7FFFu + ((ub >> 16) & 1u)) >> 16;
    return (ub << 16) | ua;
}

// --------------------------- elementwise helpers ---------------------------

__global__ void zero_kernel(float* __restrict__ p, int n) {
    int i = blockIdx.x * blockDim.x + threadIdx.x;
    if (i < n) p[i] = 0.0f;
}

__global__ void node_gather_kernel(const int* __restrict__ x,
                                   const float* __restrict__ emb,
                                   float* __restrict__ h, int N, int D) {
    int i = blockIdx.x * blockDim.x + threadIdx.x;
    if (i >= N * D) return;
    unsigned n = (unsigned)i / (unsigned)D;
    unsigned d = (unsigned)i - n * (unsigned)D;
    h[i] = emb[(unsigned)x[n] * (unsigned)D + d];
}

// message: agg[dst] += relu(h[src] + (edge_attr @ edge_W + edge_b))
__global__ void msg_kernel(const float* __restrict__ h,
                           const int* __restrict__ ei,
                           const float* __restrict__ eattr,
                           const float* __restrict__ eW,
                           const float* __restrict__ eb,
                           float* __restrict__ agg, int E, int D) {
    int i = blockIdx.x * blockDim.x + threadIdx.x;
    if (i >= E * D) return;
    unsigned e = (unsigned)i / (unsigned)D;
    unsigned d = (unsigned)i - e * (unsigned)D;
    unsigned src = (unsigned)ei[e];
    unsigned dst = (unsigned)ei[(unsigned)E + e];
    float ea = eb[d];
#pragma unroll
    for (unsigned k = 0; k < 7; ++k)
        ea += eattr[e * 7u + k] * eW[k * (unsigned)D + d];
    float m = h[src * (unsigned)D + d] + ea;
    m = m > 0.0f ? m : 0.0f;
    atomicAdd(&agg[dst * (unsigned)D + d], m);
}

// agg = (1+eps[l])*h + agg   (in place on agg)
__global__ void combine_kernel(const float* __restrict__ h,
                               const float* __restrict__ eps, int l,
                               float* __restrict__ agg, int n) {
    int i = blockIdx.x * blockDim.x + threadIdx.x;
    if (i >= n) return;
    float s = 1.0f + eps[l];
    agg[i] = s * h[i] + agg[i];
}

// per-column sums / sums of squares (batchnorm batch stats), 64 rows per block
__global__ void colstats_kernel(const float* __restrict__ X,
                                float* __restrict__ sums,
                                float* __restrict__ sumsq, int M, int D) {
    int r0 = blockIdx.x * 64;
    int rend = (r0 + 64 < M) ? (r0 + 64) : M;
    for (int cb = 0; cb < D; cb += blockDim.x) {
        int c = cb + threadIdx.x;
        if (c >= D) break;
        float s = 0.0f, sq = 0.0f;
        for (int r = r0; r < rend; ++r) {
            float v = X[(unsigned)r * (unsigned)D + (unsigned)c];
            s += v;
            sq += v * v;
        }
        atomicAdd(&sums[c], s);
        atomicAdd(&sumsq[c], sq);
    }
}

// Y = relu((X - mean) * rsqrt(var + eps) * g + be), biased batch variance
__global__ void bnrelu_kernel(const float* __restrict__ X,
                              const float* __restrict__ sums,
                              const float* __restrict__ sumsq,
                              const float* __restrict__ g,
                              const float* __restrict__ be,
                              float* __restrict__ Y, int M, int D) {
    int i = blockIdx.x * blockDim.x + threadIdx.x;
    if (i >= M * D) return;
    unsigned c = (unsigned)i % (unsigned)D;
    float invM = 1.0f / (float)M;
    float mean = sums[c] * invM;
    float var = sumsq[c] * invM - mean * mean;
    float z = (X[i] - mean) * rsqrtf(var + 1e-5f) * g[c] + be[c];
    Y[i] = z > 0.0f ? z : 0.0f;
}

__global__ void count_kernel(const int* __restrict__ batch,
                             float* __restrict__ counts, int N) {
    int i = blockIdx.x * blockDim.x + threadIdx.x;
    if (i < N) atomicAdd(&counts[batch[i]], 1.0f);
}

__global__ void poolsum_kernel(const float* __restrict__ h,
                               const int* __restrict__ batch,
                               float* __restrict__ pooled, int N, int D) {
    int i = blockIdx.x * blockDim.x + threadIdx.x;
    if (i >= N * D) return;
    unsigned n = (unsigned)i / (unsigned)D;
    unsigned d = (unsigned)i - n * (unsigned)D;
    atomicAdd(&pooled[(unsigned)batch[n] * (unsigned)D + d], h[i]);
}

__global__ void pooldiv_kernel(float* __restrict__ pooled,
                               const float* __restrict__ counts, int G, int D) {
    int i = blockIdx.x * blockDim.x + threadIdx.x;
    if (i >= G * D) return;
    unsigned gidx = (unsigned)i / (unsigned)D;
    float c = counts[gidx];
    pooled[i] = pooled[i] / (c > 1.0f ? c : 1.0f);
}

// out[g, c] = hidden[g, :] @ Wc2 + bc2
__global__ void head2_kernel(const float* __restrict__ hidden,
                             const float* __restrict__ Wc2,
                             const float* __restrict__ bc2,
                             float* __restrict__ out, int G, int H, int NC) {
    int i = blockIdx.x * blockDim.x + threadIdx.x;
    if (i >= G * NC) return;
    unsigned gidx = (unsigned)i / (unsigned)NC;
    unsigned c = (unsigned)i - gidx * (unsigned)NC;
    float acc = bc2[c];
    for (unsigned k = 0; k < (unsigned)H; ++k)
        acc += hidden[gidx * (unsigned)H + k] * Wc2[k * (unsigned)NC + c];
    out[i] = acc;
}

// --------------------------- WMMA GEMM ------------------------------------
// C[M,Nc] = A[M,K] @ W[K,Nc] + bias (optional relu). bf16 inputs, f32 accum.
// Block = 256 threads = 8 wave32s; block tile 64(M) x 32(N); K steps of 32.
// Requires K even (true for all call sites: K = 300).
__global__ __launch_bounds__(256) void gemm_bias_wmma(
    const float* __restrict__ A, const float* __restrict__ W,
    const float* __restrict__ bias, float* __restrict__ C,
    int M, int Nc, int K, int relu_flag) {
    // A tile: 64 rows x 32 k  -> 16 u32 per row (+4 pad)
    // B tile stored transposed: Bt[n][k], 32 n-rows x 16 u32 (+4 pad)
    __shared__ unsigned int Als[64 * 20];
    __shared__ unsigned int Bls[32 * 20];

    const int t = threadIdx.x;
    const int lane = t & 31;
    const int wave = t >> 5;
    const int wm = wave & 3;   // 4 row tiles of 16
    const int wn = wave >> 2;  // 2 col tiles of 16
    const int rowBase = blockIdx.y * 64;
    const int colBase = blockIdx.x * 32;
    const int hi = (lane & 16) ? 1 : 0;
    const unsigned uK = (unsigned)K, uNc = (unsigned)Nc;

    v8f acc = {};
    const int KT = (K + 31) / 32;
    for (int kt = 0; kt < KT; ++kt) {
        const int k0 = kt * 32;
        // ---- stage A tile (f32 -> packed bf16), branch-free, 32-bit math ----
#pragma unroll
        for (int it = 0; it < 4; ++it) {
            int s = t + it * 256;      // 1024 u32 slots
            int r = s >> 4, cp = s & 15;
            int ar = rowBase + r;
            int gk = k0 + 2 * cp;
            // clamp address in-bounds; load unconditionally (no exec branches)
            unsigned arc = (unsigned)(ar < M ? ar : M - 1);
            unsigned gkc = (unsigned)(gk + 1 < K ? gk : K - 2);
            float2 va = *reinterpret_cast<const float2*>(A + arc * uK + gkc);
            bool ok = (ar < M) & (gk + 1 < K);   // K even: pair all-or-nothing
            float a0 = ok ? va.x : 0.0f;
            float a1 = ok ? va.y : 0.0f;
            Als[r * 20 + cp] = pack2bf(a0, a1);
        }
        // ---- stage B tile transposed: Bt[n][k], branch-free ----
#pragma unroll
        for (int it = 0; it < 2; ++it) {
            int s = t + it * 256;      // 512 u32 slots
            int kp = s >> 5;           // 0..15
            int n = s & 31;            // 0..31
            int gn = colBase + n;
            int gk = k0 + 2 * kp;
            unsigned gnc  = (unsigned)(gn < Nc ? gn : Nc - 1);
            unsigned gk0c = (unsigned)(gk + 1 < K ? gk : K - 2);
            float w0 = W[gk0c * uNc + gnc];
            float w1 = W[(gk0c + 1u) * uNc + gnc];
            bool ok = (gn < Nc) & (gk + 1 < K);
            w0 = ok ? w0 : 0.0f;
            w1 = ok ? w1 : 0.0f;
            Bls[n * 20 + kp] = pack2bf(w0, w1);
        }
        // hint next A k-tile toward the caches (gfx1250 global_prefetch)
        if (kt + 1 < KT) {
            int pr = rowBase + (t & 63);
            unsigned prc = (unsigned)(pr < M ? pr : M - 1);
            const float* pf = A + prc * uK + (unsigned)(k0 + 32 < K ? k0 + 32 : 0);
            __builtin_prefetch(pf, 0, 0);
        }
        __syncthreads();

        // ---- build fragments per CDNA5 16-bit WMMA VGPR layouts ----
        FragU af, bf;
        {
            int m = wm * 16 + (lane & 15);
            int a0u = hi ? 4 : 0;    // elems 0..7: K 0..7 (lo lanes) / 8..15 (hi)
            int a1u = hi ? 12 : 8;   // elems 8..15: K 16..23 / 24..31
#pragma unroll
            for (int j = 0; j < 4; ++j) {
                af.u[j]     = Als[m * 20 + a0u + j];
                af.u[4 + j] = Als[m * 20 + a1u + j];
            }
            int n = wn * 16 + (lane & 15);
            int b0u = hi ? 4 : 0;    // elems 0..7: K 0..7 / 8..15; elems 8..15: K 16..23 / 24..31
#pragma unroll
            for (int j = 0; j < 8; ++j)
                bf.u[j] = Bls[n * 20 + b0u + j + (j >= 4 ? 4 : 0)];
        }
        acc = __builtin_amdgcn_wmma_f32_16x16x32_bf16(
            false, af.v, false, bf.v, (short)0, acc, false, false);
        __syncthreads();
    }

    // ---- epilogue: C layout — VGPR r: row = r + 8*hi, col = lane&15 ----
    int col = colBase + wn * 16 + (lane & 15);
    int rbase = rowBase + wm * 16 + (hi ? 8 : 0);
    if (col < Nc) {
        float bv = bias ? bias[col] : 0.0f;
#pragma unroll
        for (int r = 0; r < 8; ++r) {
            int row = rbase + r;
            if (row < M) {
                float v = acc[r] + bv;
                if (relu_flag) v = v > 0.0f ? v : 0.0f;
                C[(unsigned)row * uNc + (unsigned)col] = v;
            }
        }
    }
}

// --------------------------- launcher --------------------------------------

extern "C" void kernel_launch(void* const* d_in, const int* in_sizes, int n_in,
                              void* d_out, int out_size, void* d_ws, size_t ws_size,
                              hipStream_t stream) {
    (void)in_sizes; (void)n_in; (void)out_size; (void)ws_size;

    const int N = 20000, E = 320000, D = 300, L = 5, G = 128, NC = 6, H = 150;

    const int*   x          = (const int*)  d_in[0];
    const int*   edge_index = (const int*)  d_in[1];
    const float* edge_attr  = (const float*)d_in[2];
    const int*   batch      = (const int*)  d_in[3];
    const float* node_emb   = (const float*)d_in[4];
    const float* edge_W     = (const float*)d_in[5];
    const float* edge_b     = (const float*)d_in[6];
    const float* eps        = (const float*)d_in[7];
    const float* W1  = (const float*)d_in[8];
    const float* b1  = (const float*)d_in[9];
    const float* g1  = (const float*)d_in[10];
    const float* be1 = (const float*)d_in[11];
    const float* W2  = (const float*)d_in[12];
    const float* b2  = (const float*)d_in[13];
    const float* g2  = (const float*)d_in[14];
    const float* be2 = (const float*)d_in[15];
    const float* Wc1 = (const float*)d_in[16];
    const float* bc1 = (const float*)d_in[17];
    const float* Wc2 = (const float*)d_in[18];
    const float* bc2 = (const float*)d_in[19];
    float* out = (float*)d_out;

    // workspace layout (floats): h | agg | tmp | sums(320) | sumsq(320) |
    //                            pooled(G*D) | counts(G) | hidden(G*H)
    float* ws     = (float*)d_ws;
    float* h      = ws;
    float* agg    = ws + (size_t)N * D;
    float* tmp    = ws + 2 * (size_t)N * D;
    float* sums   = ws + 3 * (size_t)N * D;
    float* sumsq  = sums + 320;
    float* pooled = sumsq + 320;
    float* counts = pooled + (size_t)G * D;
    float* hidden = counts + G;

    const int nd = N * D;                 // 6,000,000
    const int ed = E * D;                 // 96,000,000
    const dim3 blk(256);
    const dim3 gemmGrid((D + 31) / 32, (N + 63) / 64);

    node_gather_kernel<<<(nd + 255) / 256, blk, 0, stream>>>(x, node_emb, h, N, D);

    for (int l = 0; l < L; ++l) {
        zero_kernel<<<(nd + 255) / 256, blk, 0, stream>>>(agg, nd);
        msg_kernel<<<(ed + 255) / 256, blk, 0, stream>>>(
            h, edge_index, edge_attr, edge_W, edge_b, agg, E, D);
        combine_kernel<<<(nd + 255) / 256, blk, 0, stream>>>(h, eps, l, agg, nd);

        // z @ W1 + b1 -> tmp ; BN(g1,be1) + relu in place
        gemm_bias_wmma<<<gemmGrid, blk, 0, stream>>>(
            agg, W1 + (size_t)l * D * D, b1 + (size_t)l * D, tmp, N, D, D, 0);
        zero_kernel<<<(640 + 255) / 256, blk, 0, stream>>>(sums, 640);
        colstats_kernel<<<(N + 63) / 64, blk, 0, stream>>>(tmp, sums, sumsq, N, D);
        bnrelu_kernel<<<(nd + 255) / 256, blk, 0, stream>>>(
            tmp, sums, sumsq, g1 + (size_t)l * D, be1 + (size_t)l * D, tmp, N, D);

        // tmp @ W2 + b2 -> agg ; BN(g2,be2) + relu -> h
        gemm_bias_wmma<<<gemmGrid, blk, 0, stream>>>(
            tmp, W2 + (size_t)l * D * D, b2 + (size_t)l * D, agg, N, D, D, 0);
        zero_kernel<<<(640 + 255) / 256, blk, 0, stream>>>(sums, 640);
        colstats_kernel<<<(N + 63) / 64, blk, 0, stream>>>(agg, sums, sumsq, N, D);
        bnrelu_kernel<<<(nd + 255) / 256, blk, 0, stream>>>(
            agg, sums, sumsq, g2 + (size_t)l * D, be2 + (size_t)l * D, h, N, D);
    }

    // global mean pool
    zero_kernel<<<(G * D + G + 255) / 256, blk, 0, stream>>>(pooled, G * D + G);
    count_kernel<<<(N + 255) / 256, blk, 0, stream>>>(batch, counts, N);
    poolsum_kernel<<<(nd + 255) / 256, blk, 0, stream>>>(h, batch, pooled, N, D);
    pooldiv_kernel<<<(G * D + 255) / 256, blk, 0, stream>>>(pooled, counts, G, D);

    // head: relu(pooled @ Wc1 + bc1) @ Wc2 + bc2
    gemm_bias_wmma<<<dim3((H + 31) / 32, (G + 63) / 64), blk, 0, stream>>>(
        pooled, Wc1, bc1, hidden, G, H, D, 1);
    head2_kernel<<<(G * NC + 255) / 256, blk, 0, stream>>>(
        hidden, Wc2, bc2, out, G, H, NC);
}
